// FactorizedViTBlock_82695300317702
// MI455X (gfx1250) — matrix operand order repaired and checked by reading, not compiled, not run on hardware
//
#include <hip/hip_runtime.h>
#include <cmath>

// ---------------------------------------------------------------------------
// FactorizedViTBlock for MI455X (gfx1250): bf16 WMMA GEMMs, TDM tile staging,
// double-buffered LDS, VALU axial attention.
// B=2, H=W=D=32 (65536 tokens), C=384, HEADS=6, HD=64, NNOISE=256.
// ---------------------------------------------------------------------------

typedef __bf16 bf16_t;
typedef __attribute__((ext_vector_type(16))) __bf16 v16bf;
typedef __attribute__((ext_vector_type(8)))  float  v8f;

#if defined(__has_builtin)
#  if __has_builtin(__builtin_amdgcn_tensor_load_to_lds)
#    define USE_TDM 1
#  else
#    define USE_TDM 0
#  endif
#else
#  define USE_TDM 0
#endif

__device__ __forceinline__ bf16_t f2bf(float f) {
  unsigned u = __builtin_bit_cast(unsigned, f);
  unsigned r = u + 0x7fffu + ((u >> 16) & 1u);   // round-to-nearest-even
  return __builtin_bit_cast(bf16_t, (unsigned short)(r >> 16));
}
__device__ __forceinline__ float bf2f(bf16_t h) {
  unsigned u = ((unsigned)__builtin_bit_cast(unsigned short, h)) << 16;
  return __builtin_bit_cast(float, u);
}

#if USE_TDM
typedef __attribute__((ext_vector_type(4))) unsigned tdm_u32x4;
typedef __attribute__((ext_vector_type(8))) int      tdm_i32x8;
typedef __attribute__((ext_vector_type(4))) int      tdm_i32x4;

// Flat shared-space addresses carry the LDS byte offset in the low 32 bits.
__device__ __forceinline__ unsigned lds_off_u32(const void* p) {
  return (unsigned)(unsigned long long)(size_t)p;
}

// TDM: load a 32(elem) x 128(row) bf16 tile, row length `rowlen` elems in
// global, into LDS with a 16B pad after each 64B row (LDS row stride 80B).
__device__ __forceinline__ void tdm_load_tile_32x128(const bf16_t* gsrc,
                                                     unsigned lds_byte_off,
                                                     int nrows, int rowlen) {
  unsigned long long ga = (unsigned long long)(size_t)gsrc;
  tdm_u32x4 g0;
  g0[0] = 1u;                                             // count=1 (valid), user mode
  g0[1] = lds_byte_off;                                   // lds_addr
  g0[2] = (unsigned)ga;                                   // global_addr[31:0]
  g0[3] = (unsigned)((ga >> 32) & 0x01ffffffu) | (2u << 30); // addr[56:32] | type=2
  tdm_i32x8 g1;
  // data_size=1 (2B) | pad_enable | pad_interval=3 (16 DW) | pad_amount=3 (4 DW)
  g1[0] = (int)((1u << 16) | (1u << 20) | (3u << 22) | (3u << 25));
  g1[1] = (int)(((unsigned)rowlen & 0xffffu) << 16);      // tensor_dim0 lo
  g1[2] = (int)((((unsigned)rowlen >> 16) & 0xffffu) |
                (((unsigned)nrows & 0xffffu) << 16));     // dim0 hi | dim1 lo
  g1[3] = (int)((((unsigned)nrows >> 16) & 0xffffu) |
                (32u << 16));                             // dim1 hi | tile_dim0=32
  g1[4] = 128;                                            // tile_dim1=128, tile_dim2=0
  g1[5] = rowlen;                                         // tensor_dim0_stride lo
  g1[6] = 0;
  g1[7] = 0;
  tdm_i32x4 z4 = {0, 0, 0, 0};
#if __clang_major__ >= 23
  tdm_i32x8 z8 = {0, 0, 0, 0, 0, 0, 0, 0};
  __builtin_amdgcn_tensor_load_to_lds(g0, g1, z4, z4, z8, 0);  // therock 6-arg form
#else
  __builtin_amdgcn_tensor_load_to_lds(g0, g1, z4, z4, 0);      // ROCm 7.2 5-arg form
#endif
}
#endif  // USE_TDM

// VMEM staging fallbacks (also used for the f32 A operand): 128x32 tile,
// LDS row stride 40 bf16 (80B).
__device__ __forceinline__ void stage_bf16_tile(bf16_t* dst, const bf16_t* src,
                                                int rowlen, int tid) {
  #pragma unroll
  for (int it = 0; it < 2; ++it) {
    int ch = it * 256 + tid;                // 512 chunks of 8 bf16 (16B)
    int row = ch >> 2, col = (ch & 3) << 3;
    *(uint4*)&dst[row * 40 + col] = *(const uint4*)(src + (size_t)row * rowlen + col);
  }
}
__device__ __forceinline__ void stage_f32_tile(bf16_t* dst, const float* src,
                                               int rowlen, int tid) {
  #pragma unroll
  for (int it = 0; it < 4; ++it) {
    int ch = it * 256 + tid;                // 1024 chunks of 4 floats
    int row = ch >> 3, col = (ch & 7) << 2;
    float4 d = *(const float4*)(src + (size_t)row * rowlen + col);
    bf16_t* p = &dst[row * 40 + col];
    p[0] = f2bf(d.x); p[1] = f2bf(d.y); p[2] = f2bf(d.z); p[3] = f2bf(d.w);
  }
}

// ---------------------------------------------------------------------------
// AdaLN parameter generator: params[b, 0:2304] = silu(noise[b]) @ ada_w + ada_b
// ---------------------------------------------------------------------------
__global__ void ada_kernel(const float* __restrict__ xn,
                           const float* __restrict__ aw,
                           const float* __restrict__ ab,
                           float* __restrict__ params) {
  int gidx = blockIdx.x * 256 + threadIdx.x;
  if (gidx >= 2 * 2304) return;
  int b = gidx / 2304, j = gidx % 2304;
  float acc = ab[j];
  for (int i = 0; i < 256; ++i) {
    float s = xn[b * 256 + i];
    s = s / (1.f + expf(-s));                    // silu
    acc += s * aw[(size_t)i * 2304 + j];
  }
  params[gidx] = acc;
}

// fp32 [K,N] -> bf16 [N,K] (transposed weights so GEMM B-tiles are row-major)
__global__ void cvt_t_kernel(const float* __restrict__ in, bf16_t* __restrict__ out,
                             int K, int N) {
  int idx = blockIdx.x * 256 + threadIdx.x;
  if (idx >= K * N) return;
  int k = idx / N, n = idx % N;                  // coalesced reads over n
  out[(size_t)n * K + k] = f2bf(in[idx]);
}

// ---------------------------------------------------------------------------
// LayerNorm (no affine) + AdaLN modulate, one wave32 per token -> bf16 rows
// ---------------------------------------------------------------------------
__global__ __launch_bounds__(256) void ln_mod_kernel(const float* __restrict__ x,
                                                     const float* __restrict__ params,
                                                     int shift_off, int scale_off,
                                                     bf16_t* __restrict__ out) {
  int token = blockIdx.x * 8 + (threadIdx.x >> 5);
  int lane  = threadIdx.x & 31;
  int b     = token >> 15;                       // 32768 tokens per batch
  const float* row = x + (size_t)token * 384;
  float v[12];
  float s = 0.f;
  #pragma unroll
  for (int j = 0; j < 12; ++j) { v[j] = row[lane + 32 * j]; s += v[j]; }
  #pragma unroll
  for (int m = 1; m < 32; m <<= 1) s += __shfl_xor(s, m, 32);
  float mu = s * (1.f / 384.f);
  float q = 0.f;
  #pragma unroll
  for (int j = 0; j < 12; ++j) { float d = v[j] - mu; q += d * d; }
  #pragma unroll
  for (int m = 1; m < 32; m <<= 1) q += __shfl_xor(q, m, 32);
  float rstd = rsqrtf(q * (1.f / 384.f) + 1e-5f);
  const float* pb = params + b * 2304;
  bf16_t* orow = out + (size_t)token * 384;
  #pragma unroll
  for (int j = 0; j < 12; ++j) {
    int c = lane + 32 * j;
    float y = (v[j] - mu) * rstd * (1.f + pb[scale_off + c]) + pb[shift_off + c];
    orow[c] = f2bf(y);
  }
}

// ---------------------------------------------------------------------------
// Axial attention (S=32, HD=64). One block per (b, head, fixed coords).
// qkv: bf16 [token,1152]. oacc: f32 [token,384], write (add=0) or accumulate.
// ---------------------------------------------------------------------------
__global__ __launch_bounds__(256) void attn_kernel(const bf16_t* __restrict__ qkv,
                                                   float* __restrict__ oacc,
                                                   int axis, int add) {
  __shared__ float qs[32 * 68], ks[32 * 68], vs[32 * 68], sc[32 * 33];
  int xid  = blockIdx.x;
  int head = xid % 6;
  int r    = xid / 6;
  int o1   = r & 31, o2 = (r >> 5) & 31, b = r >> 10;
  int base, stride;
  if (axis == 3)      { base = ((b * 32 + o2) * 32 + o1) * 32; stride = 1;    }  // seq over d
  else if (axis == 2) { base = (b * 32 + o2) * 1024 + o1;      stride = 32;   }  // seq over w
  else                { base = b * 32768 + o2 * 32 + o1;       stride = 1024; }  // seq over h
  int qc = head * 64, kc = 384 + head * 64, vc = 768 + head * 64;

  #pragma unroll
  for (int it = 0; it < 8; ++it) {
    int e = it * 256 + threadIdx.x;
    int s = e >> 6, c = e & 63;
    size_t t = (size_t)(base + s * stride) * 1152;
    qs[s * 68 + c] = bf2f(qkv[t + qc + c]);
    ks[s * 68 + c] = bf2f(qkv[t + kc + c]);
    vs[s * 68 + c] = bf2f(qkv[t + vc + c]);
  }
  __syncthreads();

  int i = threadIdx.x >> 3, jg = threadIdx.x & 7;  // 8 lanes per query row
  float p[4];
  #pragma unroll
  for (int jj = 0; jj < 4; ++jj) {
    int j = jg * 4 + jj;
    float d = 0.f;
    for (int c = 0; c < 64; ++c) d += qs[i * 68 + c] * ks[j * 68 + c];
    p[jj] = d * 0.125f;                            // 1/sqrt(64)
  }
  float mx = fmaxf(fmaxf(p[0], p[1]), fmaxf(p[2], p[3]));
  #pragma unroll
  for (int m = 1; m < 8; m <<= 1) mx = fmaxf(mx, __shfl_xor(mx, m, 32));
  float sum = 0.f;
  #pragma unroll
  for (int jj = 0; jj < 4; ++jj) { p[jj] = expf(p[jj] - mx); sum += p[jj]; }
  #pragma unroll
  for (int m = 1; m < 8; m <<= 1) sum += __shfl_xor(sum, m, 32);
  float inv = 1.f / sum;
  #pragma unroll
  for (int jj = 0; jj < 4; ++jj) sc[i * 33 + jg * 4 + jj] = p[jj] * inv;
  __syncthreads();

  int cg = threadIdx.x & 7;
  #pragma unroll
  for (int cc = 0; cc < 8; ++cc) {
    int c = cg * 8 + cc;
    float o = 0.f;
    for (int j = 0; j < 32; ++j) o += sc[i * 33 + j] * vs[j * 68 + c];
    size_t idx = (size_t)(base + i * stride) * 384 + head * 64 + c;
    if (add) oacc[idx] += o; else oacc[idx] = o;
  }
}

// ---------------------------------------------------------------------------
// Tiled bf16 WMMA GEMM: C[M,N] = A[M,K] @ B[K,N] + bias.
// A: [M,K] row-major (bf16, or f32 when AF32). Bt: weights pre-transposed to
// [N,K] row-major bf16, so both operand tiles are 128x32 row-major and can be
// staged by the TDM with LDS row padding. Double-buffered LDS; 8 waves, each
// owns a 32x64 output tile = 2x4 v_wmma_f32_16x16x32_bf16 per K step.
// EPI 0: store bf16. EPI 1: exact GELU -> bf16. EPI 2: resid + gate*val -> f32.
// ---------------------------------------------------------------------------
template <int EPI, bool AF32>
__global__ __launch_bounds__(256) void gemm_kernel(const void* __restrict__ Ap,
                                                   const bf16_t* __restrict__ Bt,
                                                   const float* __restrict__ bias,
                                                   const float* __restrict__ gate,
                                                   const float* __restrict__ resid,
                                                   void* __restrict__ outp,
                                                   int M, int N, int K) {
  __shared__ bf16_t As[2][128 * 40];
  __shared__ bf16_t Bs[2][128 * 40];
  const int tid  = threadIdx.x;
  const int wave = tid >> 5, lane = tid & 31;
  const int wm = wave & 3, wn = wave >> 2;        // 4x2 wave grid
  const int lrow = lane & 15, g = lane >> 4;
  const int blockM = blockIdx.x * 128, blockN = blockIdx.y * 128;

  const bf16_t* Asrc  = (const bf16_t*)Ap + (size_t)blockM * K;
  const float*  Asrcf = (const float*) Ap + (size_t)blockM * K;
  const bf16_t* Bsrc  = Bt + (size_t)blockN * K;

  auto issue = [&](int buf, int k0) {
#if USE_TDM
    if constexpr (!AF32) {
      if (wave == 0) {
        tdm_load_tile_32x128(Asrc + k0, lds_off_u32(&As[buf][0]), M, K);
        tdm_load_tile_32x128(Bsrc + k0, lds_off_u32(&Bs[buf][0]), N, K);
      }
    } else {
      stage_f32_tile(As[buf], Asrcf + k0, K, tid);
      if (wave == 0) {
        tdm_load_tile_32x128(Bsrc + k0, lds_off_u32(&Bs[buf][0]), N, K);
      }
    }
#else
    if constexpr (!AF32) stage_bf16_tile(As[buf], Asrc + k0, K, tid);
    else                 stage_f32_tile (As[buf], Asrcf + k0, K, tid);
    stage_bf16_tile(Bs[buf], Bsrc + k0, K, tid);
#endif
  };

  v8f acc[2][4] = {};
  issue(0, 0);
  int buf = 0;
  for (int k0 = 0; k0 < K; k0 += 32) {
#if USE_TDM
    if (wave == 0) __builtin_amdgcn_s_wait_tensorcnt(0);   // tiles for `buf` landed
#endif
    __syncthreads();
    if (k0 + 32 < K) issue(buf ^ 1, k0 + 32);              // overlap next tile

    const bf16_t* asb = As[buf];
    const bf16_t* bsb = Bs[buf];
    v16bf afr[2], bfr[4];
    #pragma unroll
    for (int mi = 0; mi < 2; ++mi) {
      const bf16_t* p = &asb[(wm * 32 + mi * 16 + lrow) * 40];
      #pragma unroll
      for (int v = 0; v < 8; ++v) {
        int ka = ((v >> 2) << 4) + (g << 3) + ((v & 3) << 1);  // A frag K pairs
        afr[mi][2 * v] = p[ka]; afr[mi][2 * v + 1] = p[ka + 1];
      }
    }
    #pragma unroll
    for (int ni = 0; ni < 4; ++ni) {
      const bf16_t* p = &bsb[(wn * 64 + ni * 16 + lrow) * 40];
      #pragma unroll
      for (int v = 0; v < 8; ++v) {
        int kb = ((v >> 2) << 4) + (g << 3) + ((v & 3) << 1);  // B frag K pairs
        bfr[ni][2 * v] = p[kb]; bfr[ni][2 * v + 1] = p[kb + 1];
      }
    }
    #pragma unroll
    for (int mi = 0; mi < 2; ++mi)
      #pragma unroll
      for (int ni = 0; ni < 4; ++ni)
        acc[mi][ni] = __builtin_amdgcn_wmma_f32_16x16x32_bf16(
            false, afr[mi], false, bfr[ni], (short)0, acc[mi][ni], false, false);
    __syncthreads();
    buf ^= 1;
  }

  #pragma unroll
  for (int mi = 0; mi < 2; ++mi)
    #pragma unroll
    for (int ni = 0; ni < 4; ++ni) {
      int colg = blockN + wn * 64 + ni * 16 + lrow;
      float bcol = bias[colg];
      #pragma unroll
      for (int r = 0; r < 8; ++r) {
        int rowg = blockM + wm * 32 + mi * 16 + g * 8 + r;
        float val = acc[mi][ni][r] + bcol;
        size_t idx = (size_t)rowg * N + colg;
        if constexpr (EPI == 0) {
          ((bf16_t*)outp)[idx] = f2bf(val);
        } else if constexpr (EPI == 1) {
          float ge = 0.5f * val * (1.f + erff(val * 0.70710678f));
          ((bf16_t*)outp)[idx] = f2bf(ge);
        } else {
          int bb = rowg >> 15;                              // batch from token row
          ((float*)outp)[idx] = resid[idx] + gate[bb * 2304 + colg] * val;
        }
      }
    }
}

// ---------------------------------------------------------------------------
// Launcher
// ---------------------------------------------------------------------------
extern "C" void kernel_launch(void* const* d_in, const int* in_sizes, int n_in,
                              void* d_out, int out_size, void* d_ws, size_t ws_size,
                              hipStream_t stream) {
  (void)in_sizes; (void)n_in; (void)out_size; (void)ws_size;
  const float* x      = (const float*)d_in[0];
  const float* xnoise = (const float*)d_in[1];
  const float* ada_w  = (const float*)d_in[2];
  const float* ada_b  = (const float*)d_in[3];
  const float* qkv_w  = (const float*)d_in[4];
  const float* qkv_b  = (const float*)d_in[5];
  const float* out_w  = (const float*)d_in[6];
  const float* out_b  = (const float*)d_in[7];
  const float* mlp_w1 = (const float*)d_in[8];
  const float* mlp_b1 = (const float*)d_in[9];
  const float* mlp_w2 = (const float*)d_in[10];
  const float* mlp_b2 = (const float*)d_in[11];

  char* ws = (char*)d_ws;
  size_t off = 0;
  auto alloc = [&](size_t bytes) -> void* {
    void* p = ws + off;
    off += (bytes + 255) & ~(size_t)255;
    return p;
  };
  float*  params = (float*) alloc((size_t)2 * 2304 * 4);        // AdaLN params
  bf16_t* wqkv   = (bf16_t*)alloc((size_t)1152 * 384 * 2);      // [N,K] bf16
  bf16_t* wout   = (bf16_t*)alloc((size_t)384 * 384 * 2);
  bf16_t* w1     = (bf16_t*)alloc((size_t)1536 * 384 * 2);
  bf16_t* w2     = (bf16_t*)alloc((size_t)384 * 1536 * 2);
  bf16_t* Abf    = (bf16_t*)alloc((size_t)65536 * 384 * 2);     // LN+mod rows (reused)
  bf16_t* qkvbf  = (bf16_t*)alloc((size_t)65536 * 1152 * 2);    // fused qkv
  float*  attnf  = (float*) alloc((size_t)65536 * 384 * 4);     // 3-axis attn sum
  float*  x1     = (float*) alloc((size_t)65536 * 384 * 4);     // post-attn residual
  bf16_t* hbf    = (bf16_t*)alloc((size_t)65536 * 1536 * 2);    // gelu(mlp1)

  ada_kernel<<<18, 256, 0, stream>>>(xnoise, ada_w, ada_b, params);
  cvt_t_kernel<<<(442368 + 255) / 256, 256, 0, stream>>>(qkv_w, wqkv, 384, 1152);
  cvt_t_kernel<<<(147456 + 255) / 256, 256, 0, stream>>>(out_w, wout, 384, 384);
  cvt_t_kernel<<<(589824 + 255) / 256, 256, 0, stream>>>(mlp_w1, w1, 384, 1536);
  cvt_t_kernel<<<(589824 + 255) / 256, 256, 0, stream>>>(mlp_w2, w2, 1536, 384);

  // norm1 + modulate(shift1, scale1) -> bf16
  ln_mod_kernel<<<8192, 256, 0, stream>>>(x, params, 0, 384, Abf);
  // qkv = norm_x @ qkv_w + qkv_b  (store bf16)
  gemm_kernel<0, false><<<dim3(512, 9), 256, 0, stream>>>(
      Abf, wqkv, qkv_b, nullptr, nullptr, qkvbf, 65536, 1152, 384);

  // three axial attentions, summed into attnf (f32)
  attn_kernel<<<12288, 256, 0, stream>>>(qkvbf, attnf, 3, 0);
  attn_kernel<<<12288, 256, 0, stream>>>(qkvbf, attnf, 2, 1);
  attn_kernel<<<12288, 256, 0, stream>>>(qkvbf, attnf, 1, 1);

  // x1 = x + gate1 * (attn @ out_w + out_b)   (A is f32, converted on stage)
  gemm_kernel<2, true><<<dim3(512, 3), 256, 0, stream>>>(
      attnf, wout, out_b, params + 768, x, x1, 65536, 384, 384);

  // norm2 + modulate(shift2, scale2) -> bf16
  ln_mod_kernel<<<8192, 256, 0, stream>>>(x1, params, 1152, 1536, Abf);
  // h = gelu(norm @ w1 + b1)  (store bf16)
  gemm_kernel<1, false><<<dim3(512, 12), 256, 0, stream>>>(
      Abf, w1, mlp_b1, nullptr, nullptr, hbf, 65536, 1536, 384);
  // out = x1 + gate2 * (h @ w2 + b2)
  gemm_kernel<2, false><<<dim3(512, 3), 256, 0, stream>>>(
      hbf, w2, mlp_b2, params + 1920, x1, (float*)d_out, 65536, 384, 1536);
}
